// MoEDecoder_40759239639446
// MI455X (gfx1250) — matrix operand order repaired and checked by reading, not compiled, not run on hardware
//
#include <hip/hip_runtime.h>

// ---------------- CDNA5 (gfx1250) wave32 WMMA types ----------------
typedef __attribute__((ext_vector_type(16))) __bf16 v16bf;
typedef __attribute__((ext_vector_type(8)))  __bf16 v8bf;
typedef __attribute__((ext_vector_type(8)))  float  v8f;

#define D_DIM 1024
#define H_DIM 1024
#define E_NUM 8
#define NHEAD 16
#define HD    64
#define SEQ   2048
#define BATCH 2
#define TTOK  (BATCH * SEQ)   // 4096 tokens
#define EPS   1e-5f

// Async copy of 16 bytes global -> LDS (ASYNCcnt-tracked, CDNA5 path).
__device__ __forceinline__ void async_g2l_b128(unsigned lds_off, const void* gptr) {
    unsigned long long ga = (unsigned long long)(size_t)gptr;
    asm volatile("global_load_async_to_lds_b128 %0, %1, off" :: "v"(lds_off), "v"(ga) : "memory");
}

// =============== elementwise fp32 -> bf16 convert ===============
__global__ void k_cvt_bf16(const float* __restrict__ src, __bf16* __restrict__ dst, int n) {
    int i = blockIdx.x * blockDim.x + threadIdx.x;
    int stride = gridDim.x * blockDim.x;
    for (; i < n; i += stride) dst[i] = (__bf16)src[i];
}

// =============== RMSNorm: x -> bf16(normed) ===============
__global__ __launch_bounds__(256) void k_rmsnorm_bf16(const float* __restrict__ x,
                                                      const float* __restrict__ w,
                                                      __bf16* __restrict__ out) {
    int row = blockIdx.x, tid = threadIdx.x;
    const float* xr = x + (size_t)row * D_DIM;
    float ss = 0.f;
    for (int i = tid; i < D_DIM; i += 256) { float v = xr[i]; ss += v * v; }
    for (int off = 16; off > 0; off >>= 1) ss += __shfl_xor(ss, off, 32);
    __shared__ float sred[8];
    if ((tid & 31) == 0) sred[tid >> 5] = ss;
    __syncthreads();
    float tot = 0.f;
    #pragma unroll
    for (int i = 0; i < 8; ++i) tot += sred[i];
    float inv = rsqrtf(tot / (float)D_DIM + EPS);
    for (int i = tid; i < D_DIM; i += 256)
        out[(size_t)row * D_DIM + i] = (__bf16)(xr[i] * inv * w[i]);
}

// =============== residual add + RMSNorm (fused) ===============
__global__ __launch_bounds__(256) void k_addnorm(const float* __restrict__ a,
                                                 const float* __restrict__ x,
                                                 const float* __restrict__ w,
                                                 float* __restrict__ ps_out,
                                                 __bf16* __restrict__ nb_out) {
    int row = blockIdx.x, tid = threadIdx.x;
    size_t base = (size_t)row * D_DIM;
    float ss = 0.f;
    float loc[4];
    #pragma unroll
    for (int r = 0; r < 4; ++r) {
        int i = tid + r * 256;
        float s = a[base + i] + x[base + i];
        ps_out[base + i] = s;
        loc[r] = s;
        ss += s * s;
    }
    for (int off = 16; off > 0; off >>= 1) ss += __shfl_xor(ss, off, 32);
    __shared__ float sred[8];
    if ((tid & 31) == 0) sred[tid >> 5] = ss;
    __syncthreads();
    float tot = 0.f;
    #pragma unroll
    for (int i = 0; i < 8; ++i) tot += sred[i];
    float inv = rsqrtf(tot / (float)D_DIM + EPS);
    #pragma unroll
    for (int r = 0; r < 4; ++r) {
        int i = tid + r * 256;
        nb_out[base + i] = (__bf16)(loc[r] * inv * w[i]);
    }
}

// =============== WMMA bf16 GEMM: C[M,N] = A[M,K] @ B[K,N] ===============
// Wave computes a 16x64 tile (4 accumulators, one shared A fragment -> 4 WMMA
// per A-load). 8 waves/block stack in M and share a 32x64 B tile that is
// double-buffered in LDS via GLOBAL_LOAD_ASYNC_TO_LDS_B128 (ASYNCcnt).
template <int RELU, int SCALE, int ACC, int OUTBF>
__global__ __launch_bounds__(256) void k_gemm_wmma(const __bf16* __restrict__ A,
                                                   const __bf16* __restrict__ B,
                                                   float* __restrict__ C,
                                                   __bf16* __restrict__ Cb,
                                                   const float* __restrict__ rs, int rs_stride,
                                                   int M, int N, int K) {
    __shared__ __bf16 bt[2][32 * 64];   // 2 x 4 KB
    int lane = threadIdx.x & 31;
    int wv   = threadIdx.x >> 5;
    int tile_n = blockIdx.x * 64;
    int tile_m = (blockIdx.y * 8 + wv) * 16;
    int arow  = tile_m + (lane & 15);
    int ahalf = lane >> 4;      // A: lanes 0-15 hold K 0..7/16..23, lanes 16-31 hold 8..15/24..31
    int bk    = (lane >> 4) * 16;
    int bcol  = lane & 15;

    // async stage of one 32x64 B tile: 256 threads x 16B
    auto stage = [&](int k0, int buf) {
        int t  = threadIdx.x;
        int kk = t >> 3;
        int nn = (t & 7) * 8;
        const __bf16* src = B + (size_t)(k0 + kk) * N + tile_n + nn;
        unsigned ldso = (unsigned)(size_t)&bt[buf][kk * 64 + nn];
        async_g2l_b128(ldso, src);
    };

    v8f acc[4] = {v8f{}, v8f{}, v8f{}, v8f{}};
    stage(0, 0);
    int buf = 0;
    for (int k0 = 0; k0 < K; k0 += 32) {
        bool pf = (k0 + 32) < K;
        __syncthreads();                       // reads of bt[buf^1] from prev iter done
        if (pf) {
            stage(k0 + 32, buf ^ 1);
            asm volatile("s_wait_asynccnt 0x1" ::: "memory");
        } else {
            asm volatile("s_wait_asynccnt 0x0" ::: "memory");
        }
        if (pf) __builtin_prefetch((const void*)(A + (size_t)arow * K + k0 + 32), 0, 1);
        __syncthreads();                       // all waves' async waits complete

        // A fragment: two contiguous 16B chunks per lane (ISA 16-bit A layout)
        v8bf alo = *(const v8bf*)(A + (size_t)arow * K + k0 + ahalf * 8);
        v8bf ahi = *(const v8bf*)(A + (size_t)arow * K + k0 + 16 + ahalf * 8);
        v16bf afrag;
        #pragma unroll
        for (int i = 0; i < 8; ++i) { afrag[i] = alo[i]; afrag[8 + i] = ahi[i]; }

        #pragma unroll
        for (int sub = 0; sub < 4; ++sub) {
            v16bf bfrag;                       // K rows 8h+0..7 and 16+8h+0..7 of one column
            #pragma unroll
            for (int j = 0; j < 16; ++j)
                bfrag[j] = bt[buf][(ahalf * 8 + (j & 7) + (j >> 3) * 16) * 64 + sub * 16 + bcol];
            acc[sub] = __builtin_amdgcn_wmma_f32_16x16x32_bf16(false, afrag, false, bfrag,
                                                               (short)0, acc[sub], false, false);
        }
        buf ^= 1;
    }
    // epilogue; C layout: VGPR r -> M = tile_m + r + (lane>=16)*8, N = (lane&15)
    int mbase = tile_m + (lane >> 4) * 8;
    #pragma unroll
    for (int sub = 0; sub < 4; ++sub) {
        int n = tile_n + sub * 16 + (lane & 15);
        #pragma unroll
        for (int r = 0; r < 8; ++r) {
            int m = mbase + r;
            float v = acc[sub][r];
            if (RELU)  v = v > 0.f ? v : 0.f;
            if (SCALE) v *= rs[(size_t)m * rs_stride];
            if (ACC)   v += C[(size_t)m * N + n];
            if (OUTBF) Cb[(size_t)m * N + n] = (__bf16)v;
            else       C [(size_t)m * N + n] = v;
        }
    }
}

// =============== flash attention (wave-level, WMMA, kv-step 32) ===============
// One wave per (batch, head, 16-row q tile). kv streamed in 32-row tiles:
// 2 score WMMA pairs + full-K=32 P@V WMMAs. V staged via async-to-LDS.
__global__ __launch_bounds__(128) void k_attn_wmma(const __bf16* __restrict__ q,
                                                   const __bf16* __restrict__ k,
                                                   const __bf16* __restrict__ v,
                                                   __bf16* __restrict__ o) {
    __shared__ __bf16 vstage[4][32 * HD];   // 4 KB per wave
    __shared__ __bf16 pstage[4][16 * 32];   // 1 KB per wave
    int lane = threadIdx.x & 31;
    int wv   = threadIdx.x >> 5;
    int gid  = blockIdx.x * 4 + wv;
    int qt = gid & 127;            // SEQ/16
    int h  = (gid >> 7) & 15;
    int b  = gid >> 11;
    size_t tb = (size_t)b * SEQ;
    int mrow = lane & 15;
    int hsel = lane >> 4;

    // Q fragments: 2 chunks of K=32 covering HD=64 (A-operand layout)
    v16bf qa[2];
    const __bf16* qrow = q + (tb + qt * 16 + mrow) * (size_t)D_DIM + h * HD;
    #pragma unroll
    for (int c = 0; c < 2; ++c) {
        v8bf lo = *(const v8bf*)(qrow + c * 32 + hsel * 16);
        v8bf hi = *(const v8bf*)(qrow + c * 32 + hsel * 16 + 8);
        #pragma unroll
        for (int i = 0; i < 8; ++i) { qa[c][i] = lo[i]; qa[c][8 + i] = hi[i]; }
    }

    v8f accd[4] = {v8f{}, v8f{}, v8f{}, v8f{}};
    float rmax[8], rsum[8];
    #pragma unroll
    for (int r = 0; r < 8; ++r) { rmax[r] = -3.0e38f; rsum[r] = 0.f; }

    unsigned vbase = (unsigned)(size_t)&vstage[wv][0];

    for (int kv0 = 0; kv0 <= qt * 16; kv0 += 32) {
        // ---- kick off async V staging (32 rows x 64, wave-private) ----
        #pragma unroll
        for (int half = 0; half < 2; ++half) {
            int row = mrow + half * 16;
            int kr = kv0 + row; if (kr > SEQ - 1) kr = SEQ - 1;   // masked rows: clamp
            const __bf16* src = v + (tb + kr) * (size_t)D_DIM + h * HD + hsel * 32;
            unsigned ldso = vbase + (unsigned)(row * HD + hsel * 32) * 2u;
            #pragma unroll
            for (int c2 = 0; c2 < 4; ++c2)
                async_g2l_b128(ldso + c2 * 16, src + c2 * 8);
        }

        // ---- S = Q @ K^T : two 16x16 tiles over 32 kv rows ----
        v8f s0 = {}, s1 = {};
        int kr0 = kv0 + mrow;                                   // <= SEQ-1 always
        int kr1 = kv0 + 16 + mrow; if (kr1 > SEQ - 1) kr1 = SEQ - 1;
        #pragma unroll
        for (int c = 0; c < 2; ++c) {
            const __bf16* kp0 = k + (tb + kr0) * (size_t)D_DIM + h * HD + c * 32 + hsel * 16;
            const __bf16* kp1 = k + (tb + kr1) * (size_t)D_DIM + h * HD + c * 32 + hsel * 16;
            v8bf a0 = *(const v8bf*)(kp0), a1 = *(const v8bf*)(kp0 + 8);
            v8bf b0 = *(const v8bf*)(kp1), b1 = *(const v8bf*)(kp1 + 8);
            v16bf kb0, kb1;
            #pragma unroll
            for (int i = 0; i < 8; ++i) {
                kb0[i] = a0[i]; kb0[8 + i] = a1[i];
                kb1[i] = b0[i]; kb1[8 + i] = b1[i];
            }
            s0 = __builtin_amdgcn_wmma_f32_16x16x32_bf16(false, qa[c], false, kb0,
                                                         (short)0, s0, false, false);
            s1 = __builtin_amdgcn_wmma_f32_16x16x32_bf16(false, qa[c], false, kb1,
                                                         (short)0, s1, false, false);
        }

        // ---- scale + causal mask + online softmax (rows in 16-lane groups) ----
        int ki0 = kv0 + (lane & 15);
        int ki1 = kv0 + 16 + (lane & 15);
        float alpha[8];
        #pragma unroll
        for (int r = 0; r < 8; ++r) {
            int qi = qt * 16 + r + hsel * 8;
            float a = s0[r] * 0.125f; if (ki0 > qi) a = -3.0e38f;
            float bq = s1[r] * 0.125f; if (ki1 > qi) bq = -3.0e38f;
            float mx = fmaxf(a, bq);
            #pragma unroll
            for (int off = 1; off < 16; off <<= 1) mx = fmaxf(mx, __shfl_xor(mx, off, 16));
            float mnew = fmaxf(rmax[r], mx);
            float al = __expf(rmax[r] - mnew);
            float p0 = __expf(a - mnew);
            float p1 = __expf(bq - mnew);
            float ps = p0 + p1;
            #pragma unroll
            for (int off = 1; off < 16; off <<= 1) ps += __shfl_xor(ps, off, 16);
            rsum[r] = rsum[r] * al + ps;
            rmax[r] = mnew;
            alpha[r] = al;
            s0[r] = p0; s1[r] = p1;
        }
        #pragma unroll
        for (int d = 0; d < 4; ++d)
            #pragma unroll
            for (int r = 0; r < 8; ++r) accd[d][r] *= alpha[r];

        // ---- repack P (C-layout) -> A-layout via LDS: 16 x 32 tile ----
        #pragma unroll
        for (int r = 0; r < 8; ++r) {
            pstage[wv][(r + hsel * 8) * 32 + (lane & 15)]      = (__bf16)s0[r];
            pstage[wv][(r + hsel * 8) * 32 + 16 + (lane & 15)] = (__bf16)s1[r];
        }
        // intra-wave cross-lane visibility: P writes (DScnt) + V async (ASYNCcnt)
        asm volatile("s_wait_dscnt 0x0\n\ts_wait_asynccnt 0x0" ::: "memory");

        // P as A-operand, full K=32 (halfs 0..7: K=hsel*8+j, halfs 8..15: K=16+hsel*8+j)
        v16bf pa;
        #pragma unroll
        for (int j = 0; j < 8; ++j) {
            pa[j]     = pstage[wv][mrow * 32 + hsel * 8 + j];
            pa[8 + j] = pstage[wv][mrow * 32 + 16 + hsel * 8 + j];
        }
        // O += P @ V over 4 output chunks of 16 dims (full 32x16 B operands)
        #pragma unroll
        for (int d = 0; d < 4; ++d) {
            v16bf vb;
            #pragma unroll
            for (int j = 0; j < 16; ++j)
                vb[j] = vstage[wv][(hsel * 8 + (j & 7) + (j >> 3) * 16) * HD + d * 16 + (lane & 15)];
            accd[d] = __builtin_amdgcn_wmma_f32_16x16x32_bf16(false, pa, false, vb,
                                                              (short)0, accd[d], false, false);
        }
        asm volatile("s_wait_dscnt 0x0" ::: "memory");  // LDS reads retired before next overwrite
    }
    // write O (bf16, [T, NH*HD] so reshape matches reference)
    #pragma unroll
    for (int d = 0; d < 4; ++d)
        #pragma unroll
        for (int r = 0; r < 8; ++r) {
            int m = r + hsel * 8;
            float ov = accd[d][r] / rsum[r];
            o[(tb + qt * 16 + m) * (size_t)D_DIM + h * HD + d * 16 + (lane & 15)] = (__bf16)ov;
        }
}

// =============== gating: logits, softmax, top-2, aux accumulators ===============
__global__ __launch_bounds__(256) void k_gate(const __bf16* __restrict__ post,
                                              const float* __restrict__ gw,
                                              float* __restrict__ comb,
                                              float* __restrict__ auxb) {
    int lane = threadIdx.x & 31;
    int wv   = threadIdx.x >> 5;
    int t = blockIdx.x * 8 + wv;
    if (t >= TTOK) return;
    const __bf16* xr = post + (size_t)t * D_DIM;
    float lg[E_NUM];
    #pragma unroll
    for (int e = 0; e < E_NUM; ++e) lg[e] = 0.f;
    for (int i = lane; i < D_DIM; i += 32) {
        float xv = (float)xr[i];
        #pragma unroll
        for (int e = 0; e < E_NUM; ++e) lg[e] += xv * gw[i * E_NUM + e];
    }
    #pragma unroll
    for (int e = 0; e < E_NUM; ++e)
        for (int off = 16; off > 0; off >>= 1) lg[e] += __shfl_xor(lg[e], off, 32);
    if (lane == 0) {
        float mx = lg[0];
        #pragma unroll
        for (int e = 1; e < E_NUM; ++e) mx = fmaxf(mx, lg[e]);
        float p[E_NUM], se = 0.f;
        #pragma unroll
        for (int e = 0; e < E_NUM; ++e) { p[e] = __expf(lg[e] - mx); se += p[e]; }
        #pragma unroll
        for (int e = 0; e < E_NUM; ++e) p[e] /= se;
        int i1 = 0;
        #pragma unroll
        for (int e = 1; e < E_NUM; ++e) if (p[e] > p[i1]) i1 = e;
        int i2 = (i1 == 0) ? 1 : 0;
        #pragma unroll
        for (int e = 0; e < E_NUM; ++e) if (e != i1 && p[e] > p[i2]) i2 = e;
        float s2 = p[i1] + p[i2];
        #pragma unroll
        for (int e = 0; e < E_NUM; ++e) comb[(size_t)t * E_NUM + e] = 0.f;
        comb[(size_t)t * E_NUM + i1] = p[i1] / s2;
        comb[(size_t)t * E_NUM + i2] = p[i2] / s2;
        atomicAdd(&auxb[i1], 1.f);
        atomicAdd(&auxb[i2], 1.f);
        #pragma unroll
        for (int e = 0; e < E_NUM; ++e) atomicAdd(&auxb[8 + e], p[e]);
    }
}

// =============== small helpers ===============
__global__ void k_zero(float* p, int n) {
    int i = blockIdx.x * blockDim.x + threadIdx.x;
    if (i < n) p[i] = 0.f;
}
__global__ void k_copy(const float* __restrict__ s, float* __restrict__ d, int n) {
    int i = blockIdx.x * blockDim.x + threadIdx.x;
    int stride = gridDim.x * blockDim.x;
    for (; i < n; i += stride) d[i] = s[i];
}
__global__ void k_aux_final(const float* __restrict__ auxb, float* __restrict__ out) {
    float a = 0.f;
    for (int e = 0; e < E_NUM; ++e)
        a += (auxb[e] / (float)TTOK) * (auxb[8 + e] / (float)TTOK);
    out[0] = (float)E_NUM * a;
}

// =============== orchestration ===============
extern "C" void kernel_launch(void* const* d_in, const int* in_sizes, int n_in,
                              void* d_out, int out_size, void* d_ws, size_t ws_size,
                              hipStream_t stream) {
    const float* x      = (const float*)d_in[0];   // [2,2048,1024]
    const float* pre_w  = (const float*)d_in[1];
    const float* post_w = (const float*)d_in[2];
    const float* Wq     = (const float*)d_in[3];
    const float* Wk     = (const float*)d_in[4];
    const float* Wv     = (const float*)d_in[5];
    const float* Wo     = (const float*)d_in[6];
    const float* gate_w = (const float*)d_in[7];   // [1024, 8]
    const float* W1     = (const float*)d_in[8];   // [8,1024,1024]
    const float* W2     = (const float*)d_in[9];   // [8,1024,1024]
    float* out = (float*)d_out;                    // 4194304 + 1 (aux)

    const size_t MB = 1024ull * 1024ull;
    char* ws = (char*)d_ws;
    __bf16* pre_bf    = (__bf16*)(ws + 0 * MB);     // 8 MB
    __bf16* wq_bf     = (__bf16*)(ws + 8 * MB);     // 2 MB each
    __bf16* wk_bf     = (__bf16*)(ws + 10 * MB);
    __bf16* wv_bf     = (__bf16*)(ws + 12 * MB);
    __bf16* wo_bf     = (__bf16*)(ws + 14 * MB);
    __bf16* w1_bf     = (__bf16*)(ws + 16 * MB);    // 16 MB
    __bf16* w2_bf     = (__bf16*)(ws + 32 * MB);    // 16 MB
    __bf16* q_bf      = (__bf16*)(ws + 48 * MB);    // 8 MB
    __bf16* k_bf      = (__bf16*)(ws + 56 * MB);    // 8 MB
    __bf16* v_bf      = (__bf16*)(ws + 64 * MB);    // 8 MB
    __bf16* o_bf      = (__bf16*)(ws + 72 * MB);    // 8 MB
    float*  attn_proj = (float*) (ws + 80 * MB);    // 16 MB
    float*  post_sum  = (float*) (ws + 96 * MB);    // 16 MB
    __bf16* post_bf   = (__bf16*)(ws + 112 * MB);   // 8 MB
    float*  comb      = (float*) (ws + 120 * MB);   // 128 KB
    float*  auxb      = (float*) (ws + 120 * MB + 256 * 1024); // 16 floats
    __bf16* h_bf      = (__bf16*)(ws + 121 * MB);   // 8 MB

    const int NT = TTOK;                       // 4096 tokens
    const int DD = D_DIM * D_DIM;              // 1M
    dim3 gemm_grid(D_DIM / 64, NT / 128);      // N/64 x M/128
    dim3 gemm_grid_h(H_DIM / 64, NT / 128);

    k_zero<<<1, 16, 0, stream>>>(auxb, 16);

    // weights -> bf16
    k_cvt_bf16<<<2048, 256, 0, stream>>>(Wq, wq_bf, DD);
    k_cvt_bf16<<<2048, 256, 0, stream>>>(Wk, wk_bf, DD);
    k_cvt_bf16<<<2048, 256, 0, stream>>>(Wv, wv_bf, DD);
    k_cvt_bf16<<<2048, 256, 0, stream>>>(Wo, wo_bf, DD);
    k_cvt_bf16<<<4096, 256, 0, stream>>>(W1, w1_bf, E_NUM * DD);
    k_cvt_bf16<<<4096, 256, 0, stream>>>(W2, w2_bf, E_NUM * DD);

    // pre-norm -> bf16
    k_rmsnorm_bf16<<<NT, 256, 0, stream>>>(x, pre_w, pre_bf);

    // Q/K/V projections (WMMA)
    k_gemm_wmma<0, 0, 0, 1><<<gemm_grid, 256, 0, stream>>>(pre_bf, wq_bf, nullptr, q_bf,
                                                           nullptr, 0, NT, D_DIM, D_DIM);
    k_gemm_wmma<0, 0, 0, 1><<<gemm_grid, 256, 0, stream>>>(pre_bf, wk_bf, nullptr, k_bf,
                                                           nullptr, 0, NT, D_DIM, D_DIM);
    k_gemm_wmma<0, 0, 0, 1><<<gemm_grid, 256, 0, stream>>>(pre_bf, wv_bf, nullptr, v_bf,
                                                           nullptr, 0, NT, D_DIM, D_DIM);

    // flash attention: 2*16*128 = 4096 waves, 4 waves/block
    k_attn_wmma<<<1024, 128, 0, stream>>>(q_bf, k_bf, v_bf, o_bf);

    // output projection (WMMA, fp32 out)
    k_gemm_wmma<0, 0, 0, 0><<<gemm_grid, 256, 0, stream>>>(o_bf, wo_bf, attn_proj, nullptr,
                                                           nullptr, 0, NT, D_DIM, D_DIM);

    // residual + post-norm
    k_addnorm<<<NT, 256, 0, stream>>>(attn_proj, x, post_w, post_sum, post_bf);

    // gating + aux accumulators
    k_gate<<<NT / 8, 256, 0, stream>>>(post_bf, gate_w, comb, auxb);

    // final = post_sum + MoE out: seed with residual, experts accumulate
    k_copy<<<8192, 256, 0, stream>>>(post_sum, out, NT * D_DIM);

    for (int e = 0; e < E_NUM; ++e) {
        // h = comb[:,e] * relu(post @ W1[e])  (bf16 out, scale folded pre-GEMM2)
        k_gemm_wmma<1, 1, 0, 1><<<gemm_grid_h, 256, 0, stream>>>(
            post_bf, w1_bf + (size_t)e * DD, nullptr, h_bf,
            comb + e, E_NUM, NT, H_DIM, D_DIM);
        // out += h @ W2[e]
        k_gemm_wmma<0, 0, 1, 0><<<gemm_grid, 256, 0, stream>>>(
            h_bf, w2_bf + (size_t)e * DD, out, nullptr,
            nullptr, 0, NT, D_DIM, H_DIM);
    }

    // aux loss scalar as last output element
    k_aux_final<<<1, 1, 0, stream>>>(auxb, out + (out_size - 1));
}
